// QLModel_20461224198801
// MI455X (gfx1250) — matrix operand order repaired and checked by reading, not compiled, not run on hardware
//
#include <hip/hip_runtime.h>
#include <math.h>

typedef __attribute__((ext_vector_type(16))) __bf16          v16bf;
typedef __attribute__((ext_vector_type(16))) unsigned short  v16u;
typedef __attribute__((ext_vector_type(8)))  float           v8f;

__device__ __forceinline__ unsigned short f2bf(float x) {
    unsigned int u = __float_as_uint(x);
    unsigned int r = u + 0x7FFFu + ((u >> 16) & 1u);   // round-to-nearest-even
    return (unsigned short)(r >> 16);
}

// ---------------------------------------------------------------------------
// Pack a row-major f32 weight matrix [768, N] into bf16 WMMA B-fragments.
// Layout: frag[kt][nt][lane][16]; lane<16 holds K=kt*32+{0..7,16..23} of col nt*16+lane,
// lane>=16 holds K=kt*32+{8..15,24..31} of col nt*16+(lane&15).  (ISA 7.12.2)
// ---------------------------------------------------------------------------
__global__ __launch_bounds__(256) void k_pack(const float* __restrict__ W,
                                              unsigned short* __restrict__ dst,
                                              int N) {
    int idx    = blockIdx.x * 256 + threadIdx.x;
    int i      = idx & 15;
    int lane   = (idx >> 4) & 31;
    int tile   = idx >> 9;
    int ntiles = N >> 4;
    int nt     = tile % ntiles;
    int kt     = tile / ntiles;
    int n      = nt * 16 + (lane & 15);
    int k      = kt * 32 + ((i < 8) ? (((lane >= 16) ? 8 : 0) + i) : (16 + ((lane >= 16) ? 8 : 0) + (i - 8)));
    dst[idx]   = f2bf(W[(size_t)k * N + n]);
}

// ---------------------------------------------------------------------------
// Fused: Y = X@W1 + b1 -> LayerNorm -> GELU(exact) -> dot(w2)+b2 -> logits
// grid.x = rows/32, block = 256 (8 waves). Wave w owns N-tiles w*6..w*6+5.
// Each workgroup covers TWO 16-row M-tiles and reuses each B fragment for
// both (halves total L2 B-matrix traffic vs one M-tile per workgroup).
// ---------------------------------------------------------------------------
#define LNW 772   // padded LDS row stride (avoids 64-bank conflicts)
__global__ __launch_bounds__(256) void k_mlp(const float* __restrict__ X,
                                             const unsigned short* __restrict__ Wp,
                                             const float* __restrict__ b1,
                                             const float* __restrict__ lng,
                                             const float* __restrict__ lnb,
                                             const float* __restrict__ w2,
                                             const float* __restrict__ b2,
                                             float* __restrict__ out) {
    __shared__ __align__(32) unsigned short afrag[1024];   // two 16x32 A tiles
    __shared__ float ctile[16 * LNW];
    __shared__ float psum[256];
    __shared__ float rmean[16], rrstd[16];

    const int t    = threadIdx.x;
    const int wv   = t >> 5;
    const int ln   = t & 31;
    const int row0 = blockIdx.x * 32;

    v8f acc[2][6];
#pragma unroll
    for (int mt = 0; mt < 2; ++mt)
#pragma unroll
        for (int j = 0; j < 6; ++j)
#pragma unroll
            for (int v = 0; v < 8; ++v) acc[mt][j][v] = 0.f;

    for (int kt = 0; kt < 24; ++kt) {
        __syncthreads();
        // stage two 16x32 A tiles as bf16 fragments (lane-major, ISA A layout)
        for (int idx = t; idx < 1024; idx += 256) {
            int mt = idx >> 9;
            int e  = idx & 511;
            int l  = e >> 4, i = e & 15;
            int m  = row0 + mt * 16 + (l & 15);
            int kb = (l >= 16) ? 8 : 0;
            int kl = (i < 8) ? (kb + i) : (16 + kb + (i - 8));
            afrag[idx] = f2bf(X[(size_t)m * 768 + kt * 32 + kl]);
        }
        __syncthreads();
        v16bf av0 = *reinterpret_cast<v16bf*>(&afrag[ln * 16]);
        v16bf av1 = *reinterpret_cast<v16bf*>(&afrag[512 + ln * 16]);
#pragma unroll
        for (int j = 0; j < 6; ++j) {
            const v16bf bv = *reinterpret_cast<const v16bf*>(
                &Wp[(((size_t)kt * 48 + (wv * 6 + j)) * 32 + ln) * 16]);
            acc[0][j] = __builtin_amdgcn_wmma_f32_16x16x32_bf16(
                false, av0, false, bv, (short)0, acc[0][j], false, false);
            acc[1][j] = __builtin_amdgcn_wmma_f32_16x16x32_bf16(
                false, av1, false, bv, (short)0, acc[1][j], false, false);
        }
    }

    // Epilogue: per M-tile, spill C to LDS then LayerNorm + GELU + dot(w2).
    for (int mt = 0; mt < 2; ++mt) {
        __syncthreads();
        {   // C layout: vgpr v, lane -> row v+(ln>=16?8:0), col = tile*16+(ln&15)
            int hi = (ln >= 16) ? 8 : 0;
            int cl = ln & 15;
#pragma unroll
            for (int j = 0; j < 6; ++j) {
                int col = (wv * 6 + j) * 16 + cl;
#pragma unroll
                for (int v = 0; v < 8; ++v)
                    ctile[(v + hi) * LNW + col] = acc[mt][j][v] + b1[col];
            }
        }
        __syncthreads();
        {   // 16 threads per row
            int r = t >> 4, sub = t & 15;
            float s = 0.f, sq = 0.f;
            for (int c = sub; c < 768; c += 16) {
                float v = ctile[r * LNW + c];
                s += v; sq += v * v;
            }
            psum[t] = s;
            __syncthreads();
            if (sub == 0) {
                float tot = 0.f;
                for (int q = 0; q < 16; ++q) tot += psum[r * 16 + q];
                rmean[r] = tot * (1.0f / 768.0f);
            }
            __syncthreads();
            psum[t] = sq;
            __syncthreads();
            if (sub == 0) {
                float tot = 0.f;
                for (int q = 0; q < 16; ++q) tot += psum[r * 16 + q];
                float m  = rmean[r];
                rrstd[r] = rsqrtf(tot * (1.0f / 768.0f) - m * m + 1e-5f);
            }
            __syncthreads();
            float m = rmean[r], is = rrstd[r];
            float d = 0.f;
            for (int c = sub; c < 768; c += 16) {
                float x = (ctile[r * LNW + c] - m) * is * lng[c] + lnb[c];
                float g = 0.5f * x * (1.0f + erff(x * 0.70710678f));
                d += g * w2[c];
            }
            psum[t] = d;
            __syncthreads();
            if (sub == 0) {
                float tot = 0.f;
                for (int q = 0; q < 16; ++q) tot += psum[r * 16 + q];
                out[row0 + mt * 16 + r] = tot + b2[0];
            }
        }
    }
}

// masked softmax over 256 elements per block (in place)
__global__ __launch_bounds__(256) void k_softmax256(const int* __restrict__ mask,
                                                    float* __restrict__ w) {
    __shared__ float red[256];
    int b = blockIdx.x, t = threadIdx.x;
    float v = w[b * 256 + t];
    if (mask[b * 256 + t] == 0) v = -INFINITY;
    red[t] = v; __syncthreads();
    for (int s = 128; s > 0; s >>= 1) { if (t < s) red[t] = fmaxf(red[t], red[t + s]); __syncthreads(); }
    float mx = red[0];
    __syncthreads();
    float e = (v == -INFINITY) ? 0.f : expf(v - mx);
    red[t] = e; __syncthreads();
    for (int s = 128; s > 0; s >>= 1) { if (t < s) red[t] += red[t + s]; __syncthreads(); }
    w[b * 256 + t] = e / red[0];
}

// masked softmax over groups of 10 (in place), grid=1280, block=32
__global__ void k_softmax10(const int* __restrict__ mask, float* __restrict__ w) {
    __shared__ float vals[10];
    int g = blockIdx.x, t = threadIdx.x;
    if (t < 10) {
        float v = w[g * 10 + t];
        if (mask[g * 10 + t] == 0) v = -INFINITY;
        vals[t] = v;
    }
    __syncthreads();
    if (t == 0) {
        float mx = -INFINITY;
        for (int i = 0; i < 10; ++i) mx = fmaxf(mx, vals[i]);
        float s = 0.f;
        for (int i = 0; i < 10; ++i) { float e = (vals[i] == -INFINITY) ? 0.f : expf(vals[i] - mx); vals[i] = e; s += e; }
        float inv = 1.f / s;
        for (int i = 0; i < 10; ++i) w[g * 10 + i] = vals[i] * inv;
    }
}

// gloss[g,d] = sum_l wg[g,l]*lhs_rg[g,l,d]   (grid=1280)
__global__ __launch_bounds__(256) void k_gloss(const float* __restrict__ lhs_rg,
                                               const float* __restrict__ wg,
                                               float* __restrict__ gloss) {
    int g = blockIdx.x;
    float wl[10];
#pragma unroll
    for (int l = 0; l < 10; ++l) wl[l] = wg[g * 10 + l];
    for (int d = threadIdx.x; d < 768; d += 256) {
        float a = 0.f;
#pragma unroll
        for (int l = 0; l < 10; ++l) a += wl[l] * lhs_rg[((size_t)g * 10 + l) * 768 + d];
        gloss[(size_t)g * 768 + d] = a;
    }
}

// tar[b,d] = mean over s where tar_r==1 of lhs_r[b,s,d]   (grid=128)
__global__ __launch_bounds__(256) void k_tar(const float* __restrict__ lhs_r,
                                             const int* __restrict__ tar_r,
                                             float* __restrict__ tar) {
    __shared__ float red[256];
    __shared__ float invc;
    int b = blockIdx.x, t = threadIdx.x;
    red[t] = (tar_r[b * 256 + t] == 1) ? 1.f : 0.f;
    __syncthreads();
    for (int s = 128; s > 0; s >>= 1) { if (t < s) red[t] += red[t + s]; __syncthreads(); }
    if (t == 0) invc = 1.f / red[0];
    __syncthreads();
    for (int d = t; d < 768; d += 256) {
        float a = 0.f;
        for (int s = 0; s < 256; ++s)
            if (tar_r[b * 256 + s] == 1) a += lhs_r[((size_t)b * 256 + s) * 768 + d];
        tar[b * 768 + d] = a * invc;
    }
}

// ---------------------------------------------------------------------------
// Y = normalize(X @ Wl + bl), N=32, bf16 WMMA. grid = rows/128, 8 waves,
// each wave owns its own 16-row block; A fragments built in registers.
// ---------------------------------------------------------------------------
__global__ __launch_bounds__(256) void k_proj(const float* __restrict__ X,
                                              const unsigned short* __restrict__ Wp,
                                              const float* __restrict__ bl,
                                              float* __restrict__ Y) {
    __shared__ float cts[128 * 33];
    __shared__ float pr[256];
    int t = threadIdx.x, wv = t >> 5, ln = t & 31;
    int rowbase = (blockIdx.x * 8 + wv) * 16;
    v8f acc0, acc1;
#pragma unroll
    for (int v = 0; v < 8; ++v) { acc0[v] = 0.f; acc1[v] = 0.f; }
    int m  = rowbase + (ln & 15);
    int kb = (ln >= 16) ? 8 : 0;
    for (int kt = 0; kt < 24; ++kt) {
        const float* xp = X + (size_t)m * 768 + kt * 32;
        v16u uv;
#pragma unroll
        for (int i = 0; i < 8; ++i) uv[i]     = f2bf(xp[kb + i]);
#pragma unroll
        for (int i = 0; i < 8; ++i) uv[8 + i] = f2bf(xp[16 + kb + i]);
        v16bf av = __builtin_bit_cast(v16bf, uv);
        const v16bf bv0 = *reinterpret_cast<const v16bf*>(&Wp[(((size_t)kt * 2 + 0) * 32 + ln) * 16]);
        const v16bf bv1 = *reinterpret_cast<const v16bf*>(&Wp[(((size_t)kt * 2 + 1) * 32 + ln) * 16]);
        acc0 = __builtin_amdgcn_wmma_f32_16x16x32_bf16(false, av, false, bv0, (short)0, acc0, false, false);
        acc1 = __builtin_amdgcn_wmma_f32_16x16x32_bf16(false, av, false, bv1, (short)0, acc1, false, false);
    }
    {
        int hi = (ln >= 16) ? 8 : 0, cl = ln & 15;
#pragma unroll
        for (int v = 0; v < 8; ++v) {
            cts[(wv * 16 + v + hi) * 33 + cl]      = acc0[v] + bl[cl];
            cts[(wv * 16 + v + hi) * 33 + 16 + cl] = acc1[v] + bl[16 + cl];
        }
    }
    __syncthreads();
    {   // L2 normalize each of the 128 rows (2 threads/row)
        int r = t >> 1, h = t & 1;
        float s = 0.f;
        for (int c = h * 16; c < h * 16 + 16; ++c) { float v = cts[r * 33 + c]; s += v * v; }
        pr[t] = s;
    }
    __syncthreads();
    {
        int r = t >> 1, h = t & 1;
        float n   = sqrtf(pr[r * 2] + pr[r * 2 + 1]);
        float inv = 1.f / fmaxf(n, 1e-12f);
        size_t o  = (size_t)(blockIdx.x * 128 + r) * 32;
        for (int c = h * 16; c < h * 16 + 16; ++c) Y[o + c] = cts[r * 33 + c] * inv;
    }
}

// gw[b,g] = softmax_g(mask( dot(pooler_rg[b,g,:], pooler_r[b,:]) ))  (grid=128)
__global__ __launch_bounds__(256) void k_gw(const float* __restrict__ pooler_rg,
                                            const float* __restrict__ pooler_r,
                                            const int* __restrict__ gloss_att,
                                            float* __restrict__ gw) {
    __shared__ float red[256];
    __shared__ float res[10];
    int b = blockIdx.x, t = threadIdx.x;
    for (int g = 0; g < 10; ++g) {
        float s = 0.f;
        for (int d = t; d < 768; d += 256)
            s += pooler_rg[((size_t)b * 10 + g) * 768 + d] * pooler_r[(size_t)b * 768 + d];
        red[t] = s; __syncthreads();
        for (int st = 128; st > 0; st >>= 1) { if (t < st) red[t] += red[t + st]; __syncthreads(); }
        if (t == 0) res[g] = red[0];
        __syncthreads();
    }
    if (t == 0) {
        float mx = -INFINITY;
        for (int g = 0; g < 10; ++g) {
            if (gloss_att[b * 10 + g] == 0) res[g] = -INFINITY;
            mx = fmaxf(mx, res[g]);
        }
        float s = 0.f;
        for (int g = 0; g < 10; ++g) { float e = (res[g] == -INFINITY) ? 0.f : expf(res[g] - mx); res[g] = e; s += e; }
        for (int g = 0; g < 10; ++g) gw[b * 10 + g] = res[g] / s;
    }
}

// context_density[b] = sum_s w_s conj(x_s) x_s^T, x scaled by sqrt(w). grid=128
__global__ __launch_bounds__(256) void k_density_ctx(const float* __restrict__ ctx,
                                                     const float* __restrict__ cp,
                                                     const int* __restrict__ tar_r,
                                                     const float* __restrict__ wc,
                                                     float* __restrict__ Dre,
                                                     float* __restrict__ Dim) {
    __shared__ float xr[64 * 32], xi[64 * 32];
    int b = blockIdx.x, t = threadIdx.x;
    int p0 = t * 4;
    float rr[4] = {0, 0, 0, 0}, ii[4] = {0, 0, 0, 0};
    for (int c0 = 0; c0 < 256; c0 += 64) {
        __syncthreads();
        for (int e = t; e < 2048; e += 256) {
            int sl = e >> 5, col = e & 31;
            int srow = b * 256 + c0 + sl;
            float sw = sqrtf(wc[srow]);
            xr[e] = ctx[(size_t)srow * 32 + col] * sw;
            xi[e] = cp[tar_r[srow] * 32 + col] * sw;
        }
        __syncthreads();
        for (int s = 0; s < 64; ++s) {
#pragma unroll
            for (int q = 0; q < 4; ++q) {
                int p = p0 + q, i = p >> 5, j = p & 31;
                float ari = xr[s * 32 + i], aii = xi[s * 32 + i];
                float brj = xr[s * 32 + j], bij = xi[s * 32 + j];
                rr[q] += ari * brj + aii * bij;      // Re(conj(xi)*xj)
                ii[q] += ari * bij - aii * brj;      // Im(conj(xi)*xj)
            }
        }
    }
#pragma unroll
    for (int q = 0; q < 4; ++q) {
        Dre[(size_t)b * 1024 + p0 + q] = rr[q];
        Dim[(size_t)b * 1024 + p0 + q] = ii[q];
    }
}

// gloss_density[b] over 10 terms.  grid=128
__global__ __launch_bounds__(256) void k_density_gloss(const float* __restrict__ glP,
                                                       const float* __restrict__ gp,
                                                       const float* __restrict__ gw,
                                                       float* __restrict__ Dre,
                                                       float* __restrict__ Dim) {
    __shared__ float xr[320], xi[320];
    int b = blockIdx.x, t = threadIdx.x;
    for (int e = t; e < 320; e += 256) {
        int g = e >> 5, col = e & 31;
        float sw = sqrtf(gw[b * 10 + g]);
        xr[e] = glP[((size_t)b * 10 + g) * 32 + col] * sw;
        xi[e] = gp[g * 32 + col] * sw;
    }
    __syncthreads();
    int p0 = t * 4;
    float rr[4] = {0, 0, 0, 0}, ii[4] = {0, 0, 0, 0};
    for (int g = 0; g < 10; ++g) {
#pragma unroll
        for (int q = 0; q < 4; ++q) {
            int p = p0 + q, i = p >> 5, j = p & 31;
            float ari = xr[g * 32 + i], aii = xi[g * 32 + i];
            float brj = xr[g * 32 + j], bij = xi[g * 32 + j];
            rr[q] += ari * brj + aii * bij;
            ii[q] += ari * bij - aii * brj;
        }
    }
#pragma unroll
    for (int q = 0; q < 4; ++q) {
        Dre[(size_t)b * 1024 + p0 + q] = rr[q];
        Dim[(size_t)b * 1024 + p0 + q] = ii[q];
    }
}

// builds spv/mip = stack(|D - t t^T|, D @ (t t^T)); D@(t t^T) = (D t) t^T.
__global__ __launch_bounds__(256) void k_mix(const float* __restrict__ cDr,
                                             const float* __restrict__ cDi,
                                             const float* __restrict__ gDr,
                                             const float* __restrict__ gDi,
                                             const float* __restrict__ tarP,
                                             float* __restrict__ spvR, float* __restrict__ spvI,
                                             float* __restrict__ mipR, float* __restrict__ mipI) {
    __shared__ float tp[32];
    __shared__ float cvr[32], cvi[32], gvr[32], gvi[32];
    int b = blockIdx.x, t = threadIdx.x;
    if (t < 32) tp[t] = tarP[b * 32 + t];
    __syncthreads();
    if (t < 32) {
        float ar = 0, ai = 0, br = 0, bi = 0;
        for (int k = 0; k < 32; ++k) {
            float tk = tp[k];
            ar += cDr[(size_t)b * 1024 + t * 32 + k] * tk;
            ai += cDi[(size_t)b * 1024 + t * 32 + k] * tk;
            br += gDr[(size_t)b * 1024 + t * 32 + k] * tk;
            bi += gDi[(size_t)b * 1024 + t * 32 + k] * tk;
        }
        cvr[t] = ar; cvi[t] = ai; gvr[t] = br; gvi[t] = bi;
    }
    __syncthreads();
    for (int e = t; e < 1024; e += 256) {
        int i = e >> 5, j = e & 31;
        float td = tp[i] * tp[j];
        size_t base = (size_t)b * 2048;
        float cr = cDr[(size_t)b * 1024 + e], ci = cDi[(size_t)b * 1024 + e];
        float gr = gDr[(size_t)b * 1024 + e], gi = gDi[(size_t)b * 1024 + e];
        spvR[base + e] = sqrtf((cr - td) * (cr - td) + ci * ci);  spvI[base + e] = 0.f;
        mipR[base + e] = sqrtf((gr - td) * (gr - td) + gi * gi);  mipI[base + e] = 0.f;
        spvR[base + 1024 + e] = cvr[i] * tp[j];  spvI[base + 1024 + e] = cvi[i] * tp[j];
        mipR[base + 1024 + e] = gvr[i] * tp[j];  mipI[base + 1024 + e] = gvi[i] * tp[j];
    }
}

// complex BN (whitening) over (b,h,w) per channel; grid=2 channels, in-place
__global__ __launch_bounds__(256) void k_bn(float* __restrict__ xR, float* __restrict__ xI,
                                            const float* __restrict__ w,
                                            const float* __restrict__ bprm) {
    __shared__ float red[256];
    __shared__ float st[5];
    int c = blockIdx.x, t = threadIdx.x;
    float sr = 0, si = 0, srr = 0, sii = 0, sri = 0;
    for (int e = t; e < 131072; e += 256) {
        int b = e >> 10, p = e & 1023;
        size_t idx = (size_t)b * 2048 + c * 1024 + p;
        float r = xR[idx], i = xI[idx];
        sr += r; si += i; srr += r * r; sii += i * i; sri += r * i;
    }
    float part[5] = {sr, si, srr, sii, sri};
#pragma unroll
    for (int q = 0; q < 5; ++q) {
        red[t] = part[q]; __syncthreads();
        for (int s = 128; s > 0; s >>= 1) { if (t < s) red[t] += red[t + s]; __syncthreads(); }
        if (t == 0) st[q] = red[0];
        __syncthreads();
    }
    if (t == 0) {
        const float n = 131072.f;
        float mr = st[0] / n, mi = st[1] / n;
        float Crr = st[2] / n - mr * mr + 1e-5f;
        float Cii = st[3] / n - mi * mi + 1e-5f;
        float Cri = st[4] / n - mr * mi;
        float s   = sqrtf(Crr * Cii - Cri * Cri);
        float tt  = sqrtf(Crr + Cii + 2.f * s);
        float ist = 1.f / (s * tt);
        st[0] = mr; st[1] = mi;
        st[2] = (Cii + s) * ist;   // Rrr
        st[3] = (Crr + s) * ist;   // Rii
        st[4] = -Cri * ist;        // Rri
    }
    __syncthreads();
    float mr = st[0], mi = st[1], Rrr = st[2], Rii = st[3], Rri = st[4];
    float w0 = w[c * 3 + 0], w1 = w[c * 3 + 1], w2v = w[c * 3 + 2];
    float b0 = bprm[c * 2 + 0], b1v = bprm[c * 2 + 1];
    for (int e = t; e < 131072; e += 256) {
        int b = e >> 10, p = e & 1023;
        size_t idx = (size_t)b * 2048 + c * 1024 + p;
        float r = xR[idx] - mr, i = xI[idx] - mi;
        float zr = Rrr * r + Rri * i;
        float zi = Rii * i + Rri * r;
        xR[idx] = w0 * zr + w2v * zi + b0;
        xI[idx] = w2v * zr + w1 * zi + b1v;
    }
}

// complex conv 2->32, 3x3 VALID, 32x32 -> 30x30.  grid=128 (per batch)
__global__ __launch_bounds__(256) void k_conv(const float* __restrict__ xR,
                                              const float* __restrict__ xI,
                                              const float* __restrict__ wr,
                                              const float* __restrict__ wi,
                                              const float* __restrict__ br,
                                              const float* __restrict__ bi,
                                              float* __restrict__ outR,
                                              float* __restrict__ outI, int outOff) {
    __shared__ float sxr[2048], sxi[2048];
    __shared__ float swr[576], swi[576], sbr[32], sbi[32];
    int b = blockIdx.x, t = threadIdx.x;
    for (int e = t; e < 2048; e += 256) { sxr[e] = xR[(size_t)b * 2048 + e]; sxi[e] = xI[(size_t)b * 2048 + e]; }
    for (int e = t; e < 576; e += 256) { swr[e] = wr[e]; swi[e] = wi[e]; }
    if (t < 32) { sbr[t] = br[t]; sbi[t] = bi[t]; }
    __syncthreads();
    for (int o = t; o < 28800; o += 256) {
        int co = o / 900, p = o % 900, oy = p / 30, ox = p % 30;
        float aRR = 0, aII = 0, aIR = 0, aRI = 0;
#pragma unroll
        for (int ci = 0; ci < 2; ++ci)
#pragma unroll
            for (int ky = 0; ky < 3; ++ky)
#pragma unroll
                for (int kx = 0; kx < 3; ++kx) {
                    int ii  = ci * 1024 + (oy + ky) * 32 + (ox + kx);
                    int wix = ((co * 2 + ci) * 3 + ky) * 3 + kx;
                    float r = sxr[ii], im = sxi[ii];
                    float wR = swr[wix], wI = swi[wix];
                    aRR += r * wR; aII += im * wI; aIR += im * wR; aRI += r * wI;
                }
        size_t oidx = (size_t)b * 57600 + outOff + o;
        outR[oidx] = aRR - aII + sbr[co] - sbi[co];
        outI[oidx] = aIR + aRI + sbr[co] + sbi[co];
    }
}

// complex linear 57600 -> 2 then abs.  grid=128
__global__ __launch_bounds__(256) void k_fc(const float* __restrict__ fR,
                                            const float* __restrict__ fI,
                                            const float* __restrict__ wr,
                                            const float* __restrict__ wi,
                                            const float* __restrict__ br,
                                            const float* __restrict__ bi,
                                            float* __restrict__ out) {
    __shared__ float red[256];
    __shared__ float res[8];
    int b = blockIdx.x, t = threadIdx.x;
    float part[8] = {0, 0, 0, 0, 0, 0, 0, 0};
    for (int k = t; k < 57600; k += 256) {
        float r = fR[(size_t)b * 57600 + k], im = fI[(size_t)b * 57600 + k];
#pragma unroll
        for (int o = 0; o < 2; ++o) {
            float wR = wr[o * 57600 + k], wI = wi[o * 57600 + k];
            part[o * 4 + 0] += r * wR;
            part[o * 4 + 1] += im * wI;
            part[o * 4 + 2] += im * wR;
            part[o * 4 + 3] += r * wI;
        }
    }
#pragma unroll
    for (int q = 0; q < 8; ++q) {
        red[t] = part[q]; __syncthreads();
        for (int s = 128; s > 0; s >>= 1) { if (t < s) red[t] += red[t + s]; __syncthreads(); }
        if (t == 0) res[q] = red[0];
        __syncthreads();
    }
    if (t < 2) {
        int o = t;
        float oR = res[o * 4 + 0] - res[o * 4 + 1] + br[o] - bi[o];
        float oI = res[o * 4 + 2] + res[o * 4 + 3] + br[o] + bi[o];
        out[b * 2 + o] = sqrtf(oR * oR + oI * oI);
    }
}

extern "C" void kernel_launch(void* const* d_in, const int* in_sizes, int n_in,
                              void* d_out, int out_size, void* d_ws, size_t ws_size,
                              hipStream_t stream) {
    (void)in_sizes; (void)n_in; (void)out_size; (void)ws_size;
    const float* lhs_r     = (const float*)d_in[0];
    const float* pooler_r  = (const float*)d_in[1];
    const float* lhs_rg    = (const float*)d_in[2];
    const float* pooler_rg = (const float*)d_in[3];
    const int*   att_r     = (const int*)d_in[4];
    const int*   att_rg    = (const int*)d_in[5];
    const int*   tar_r     = (const int*)d_in[6];
    const int*   gloss_att = (const int*)d_in[7];
    // params: pytree-flattened, dict keys sorted alphabetically
    const float* at_b1  = (const float*)d_in[8];
    const float* at_b2  = (const float*)d_in[9];
    const float* at_lnb = (const float*)d_in[10];
    const float* at_lng = (const float*)d_in[11];
    const float* at_w1  = (const float*)d_in[12];
    const float* at_w2  = (const float*)d_in[13];
    const float* cm_bi  = (const float*)d_in[14];
    const float* cm_br  = (const float*)d_in[15];
    const float* cm_wi  = (const float*)d_in[16];
    const float* cm_wr  = (const float*)d_in[17];
    const float* cs_bi  = (const float*)d_in[18];
    const float* cs_br  = (const float*)d_in[19];
    const float* cs_wi  = (const float*)d_in[20];
    const float* cs_wr  = (const float*)d_in[21];
    const float* cp     = (const float*)d_in[22];
    const float* fc_bi  = (const float*)d_in[23];
    const float* fc_br  = (const float*)d_in[24];
    const float* fc_wi  = (const float*)d_in[25];
    const float* fc_wr  = (const float*)d_in[26];
    const float* gp     = (const float*)d_in[27];
    const float* nm_b   = (const float*)d_in[28];
    const float* nm_w   = (const float*)d_in[29];
    const float* ns_b   = (const float*)d_in[30];
    const float* ns_w   = (const float*)d_in[31];
    const float* pl_b1  = (const float*)d_in[32];
    const float* pl_b2  = (const float*)d_in[33];
    const float* pl_lnb = (const float*)d_in[34];
    const float* pl_lng = (const float*)d_in[35];
    const float* pl_w1  = (const float*)d_in[36];
    const float* pl_w2  = (const float*)d_in[37];
    const float* rl_b   = (const float*)d_in[38];
    const float* rl_w   = (const float*)d_in[39];

    char* ws = (char*)d_ws;
    size_t off = 0;
    auto alloc = [&](size_t bytes) -> char* {
        char* p = ws + off;
        off = (off + bytes + 255) & ~(size_t)255;
        return p;
    };
    unsigned short* wpA = (unsigned short*)alloc(589824 * 2);   // packed atten.w1
    unsigned short* wpP = (unsigned short*)alloc(589824 * 2);   // packed pool.w1
    unsigned short* wpL = (unsigned short*)alloc(24576 * 2);    // packed realliner.w
    float* w_c   = (float*)alloc(32768 * 4);      // logits -> softmax weights [B,S]
    float* w_g   = (float*)alloc(12800 * 4);      // [B*NG, GL]
    float* gloss = (float*)alloc(1280 * 768 * 4);
    float* tar   = (float*)alloc(128 * 768 * 4);
    float* ctxP  = (float*)alloc((size_t)32768 * 32 * 4);
    float* glP   = (float*)alloc(1280 * 32 * 4);
    float* tarP  = (float*)alloc(128 * 32 * 4);
    float* gw    = (float*)alloc(1280 * 4);
    float* cDr   = (float*)alloc((size_t)128 * 1024 * 4);
    float* cDi   = (float*)alloc((size_t)128 * 1024 * 4);
    float* gDr   = (float*)alloc((size_t)128 * 1024 * 4);
    float* gDi   = (float*)alloc((size_t)128 * 1024 * 4);
    float* spvR  = (float*)alloc((size_t)128 * 2048 * 4);
    float* spvI  = (float*)alloc((size_t)128 * 2048 * 4);
    float* mipR  = (float*)alloc((size_t)128 * 2048 * 4);
    float* mipI  = (float*)alloc((size_t)128 * 2048 * 4);
    float* finR  = (float*)alloc((size_t)128 * 57600 * 4);
    float* finI  = (float*)alloc((size_t)128 * 57600 * 4);
    float* out   = (float*)d_out;

    // 1) pack weights to bf16 WMMA fragments
    k_pack<<<2304, 256, 0, stream>>>(at_w1, wpA, 768);
    k_pack<<<2304, 256, 0, stream>>>(pl_w1, wpP, 768);
    k_pack<<<96, 256, 0, stream>>>(rl_w, wpL, 32);
    // 2) attention-pool MLPs (WMMA GEMM + LN + GELU + dot), 32 rows/WG
    k_mlp<<<1024, 256, 0, stream>>>(lhs_r, wpA, at_b1, at_lng, at_lnb, at_w2, at_b2, w_c);
    k_mlp<<<400, 256, 0, stream>>>(lhs_rg, wpP, pl_b1, pl_lng, pl_lnb, pl_w2, pl_b2, w_g);
    // 3) masked softmaxes
    k_softmax256<<<128, 256, 0, stream>>>(att_r, w_c);
    k_softmax10<<<1280, 32, 0, stream>>>(att_rg, w_g);
    // 4) pooled gloss + masked-mean tar
    k_gloss<<<1280, 256, 0, stream>>>(lhs_rg, w_g, gloss);
    k_tar<<<128, 256, 0, stream>>>(lhs_r, tar_r, tar);
    // 5) realliner projections + L2 normalize (WMMA)
    k_proj<<<256, 256, 0, stream>>>(lhs_r, wpL, rl_b, ctxP);
    k_proj<<<10, 256, 0, stream>>>(gloss, wpL, rl_b, glP);
    k_proj<<<1, 256, 0, stream>>>(tar, wpL, rl_b, tarP);
    // 6) gloss weights (bmm + masked softmax over NG)
    k_gw<<<128, 256, 0, stream>>>(pooler_rg, pooler_r, gloss_att, gw);
    // 7) density matrices
    k_density_ctx<<<128, 256, 0, stream>>>(ctxP, cp, tar_r, w_c, cDr, cDi);
    k_density_gloss<<<128, 256, 0, stream>>>(glP, gp, gw, gDr, gDi);
    // 8) stack |D - T| and D@T into spv/mip
    k_mix<<<128, 256, 0, stream>>>(cDr, cDi, gDr, gDi, tarP, spvR, spvI, mipR, mipI);
    // 9) complex batch norm (whitening), in place
    k_bn<<<2, 256, 0, stream>>>(spvR, spvI, ns_w, ns_b);
    k_bn<<<2, 256, 0, stream>>>(mipR, mipI, nm_w, nm_b);
    // 10) complex convs -> concatenated `final`
    k_conv<<<128, 256, 0, stream>>>(spvR, spvI, cs_wr, cs_wi, cs_br, cs_bi, finR, finI, 0);
    k_conv<<<128, 256, 0, stream>>>(mipR, mipI, cm_wr, cm_wi, cm_br, cm_bi, finR, finI, 28800);
    // 11) complex fc + abs -> output [B,2]
    k_fc<<<128, 256, 0, stream>>>(finR, finI, fc_wr, fc_wi, fc_br, fc_bi, out);
}
